// HierarchicalAttention_51479478010207
// MI455X (gfx1250) — matrix operand order repaired and checked by reading, not compiled, not run on hardware
//
#include <hip/hip_runtime.h>
#include <hip/hip_bf16.h>

typedef __attribute__((ext_vector_type(16))) _Float16 v16h;
typedef __attribute__((ext_vector_type(8)))  _Float16 v8h;
typedef __attribute__((ext_vector_type(8)))  float    v8f;
typedef __attribute__((ext_vector_type(4)))  float    v4f;

#define BB 8
#define TT 64
#define II 256
#define JJ 256
#define DD 256
#define KD 64
#define VD 64
#define NSEQ (BB * TT)              // 512
#define SCALE_PER_SEQ   0.125f      // 1/sqrt(64)
#define SCALE_OVER_SEQS 0.125f

#define WMMA_F16(a, b, c) \
  __builtin_amdgcn_wmma_f32_16x16x32_f16(false, (a), false, (b), (short)0, (c), false, false)

// ---- Fragment loaders (CDNA5 ISA 7.12.2 layouts, wave32) ----
// A 16x32 f16: lane holds row M=(L&15); elems e=0..7 -> k=k0+8*lh+e,
//              e=8..15 -> k=k0+8*lh+16+(e-8). Pass p = row + k0 + 8*lh.
__device__ __forceinline__ v16h afrag_f16(const _Float16* p) {
  v8h lo = *(const v8h*)p;
  v8h hi = *(const v8h*)(p + 16);
  return __builtin_shufflevector(lo, hi, 0,1,2,3,4,5,6,7,8,9,10,11,12,13,14,15);
}
// B 32x16 f16: lane holds col N=(L&15); elems e=0..7 -> k=k0+8*lh+e,
// e=8..15 -> k=k0+8*lh+16+(e-8). Pass p = col*stride + k0 + 8*lh.
__device__ __forceinline__ v16h bfrag_f16(const _Float16* p) {
  v8h lo = *(const v8h*)p;
  v8h hi = *(const v8h*)(p + 16);
  return __builtin_shufflevector(lo, hi, 0,1,2,3,4,5,6,7,8,9,10,11,12,13,14,15);
}
// A-fragment from f32 source (vector loads + packed cvt)
__device__ __forceinline__ v8h cvt8(const float* __restrict__ p) {
  v4f a = *(const v4f*)p;
  v4f b = *(const v4f*)(p + 4);
  v8h r;
  r[0]=(_Float16)a[0]; r[1]=(_Float16)a[1]; r[2]=(_Float16)a[2]; r[3]=(_Float16)a[3];
  r[4]=(_Float16)b[0]; r[5]=(_Float16)b[1]; r[6]=(_Float16)b[2]; r[7]=(_Float16)b[3];
  return r;
}
__device__ __forceinline__ v16h afrag_f32(const float* __restrict__ p) {
  v8h lo = cvt8(p);
  v8h hi = cvt8(p + 16);
  return __builtin_shufflevector(lo, hi, 0,1,2,3,4,5,6,7,8,9,10,11,12,13,14,15);
}

// ============================================================
// Kernel 1: projection Y = X(rows x 256) @ W(256 x 64) + bias
//   MODE 0: out[row][n]  f16 row-major  (used for Q and K)
//   MODE 1: out[(row>>8)*64 + n][row&255]  (per-seq transposed, for V^T)
// One wave: 16 rows x 64 cols. Workgroup (8 waves): 128 rows.
// W staged in LDS transposed [n][k] so B-fragments are contiguous.
// ============================================================
template <int MODE>
__global__ __launch_bounds__(256) void proj_kernel(
    const float* __restrict__ X, const float* __restrict__ W,
    const float* __restrict__ bias, _Float16* __restrict__ out) {
  __shared__ _Float16 Wt[KD * DD];  // [n][k], 32 KB
  int tid = threadIdx.x;
  for (int idx = tid; idx < DD * KD; idx += 256) {
    int d = idx >> 6, n = idx & 63;           // W row-major read, coalesced
    Wt[n * DD + d] = (_Float16)W[idx];
  }
  __syncthreads();

  int wave = tid >> 5, lane = tid & 31;
  int lhalf = lane >> 4, l15 = lane & 15;
  long rowBase = (long)blockIdx.x * 128 + wave * 16;
  const float* rowp = X + (rowBase + l15) * DD;

  for (int nt = 0; nt < 4; ++nt) {
    int col = nt * 16 + l15;
    const _Float16* wp = &Wt[col * DD];
    v8f acc = {};
#pragma unroll
    for (int k0 = 0; k0 < DD; k0 += 32) {
      v16h a   = afrag_f32(rowp + k0 + 8 * lhalf);
      v16h bfr = bfrag_f16(wp + k0 + 8 * lhalf);
      acc = WMMA_F16(a, bfr, acc);
    }
    float bv = bias[col];
    if (MODE == 0) {
#pragma unroll
      for (int r = 0; r < 8; ++r)
        out[(rowBase + r + 8 * lhalf) * KD + col] = (_Float16)(acc[r] + bv);
    } else {
      long seq = rowBase >> 8;
      int  jj  = (int)(rowBase & 255) + 8 * lhalf;   // rows r -> jj+r contiguous
      v8h pk;
#pragma unroll
      for (int r = 0; r < 8; ++r) pk[r] = (_Float16)(acc[r] + bv);
      *(v8h*)&out[(seq * KD + col) * (long)JJ + jj] = pk;  // one b128 store
    }
  }
}

// ============================================================
// Kernel 2: per-(b,t) fused attention.
//   S = Q[b] (256x64) @ K[b,t]^T        -> LDS f32 (K stored [j][kd])
//   row softmax(SCALE*S); raw row-max -> SMAX; normalized P f16 in LDS
//   PSR[b,t] = P @ V[b,t]               (V stored transposed [vd][j])
// 8 waves x 32 rows; each wave owns its rows -> no block barriers.
// Dynamic LDS 256 KB; P f16 packed into first half of each row's own
// 1 KB slot (stride 512 f16) -> no cross-wave aliasing.
// ============================================================
__global__ __launch_bounds__(256) void attn_kernel(
    const _Float16* __restrict__ Qh, const _Float16* __restrict__ Kn,
    const _Float16* __restrict__ Vt, float* __restrict__ PSR,
    float* __restrict__ SMAX) {
  extern __shared__ char smem[];
  float*    Sf = (float*)smem;
  _Float16* Sh = (_Float16*)smem;

  int seq = blockIdx.x;        // seq = b*64 + t
  int b   = seq >> 6;
  int tid = threadIdx.x, wave = tid >> 5, lane = tid & 31;
  int lhalf = lane >> 4, l15 = lane & 15;
  int rbase = wave * 32;

  const _Float16* Qb = Qh + (long)b * II * KD;
  const _Float16* Kp = Kn + (long)seq * JJ * KD;   // [j][kd]
  const _Float16* Vp = Vt + (long)seq * VD * JJ;   // [vd][j]

  // ---- Phase 1: S = Q K^T (raw scores, f32 in LDS) ----
  for (int mt = 0; mt < 2; ++mt) {
    int r0 = rbase + mt * 16;
    const _Float16* qrow = Qb + (r0 + l15) * KD + 8 * lhalf;
    for (int nt = 0; nt < 16; ++nt) {
      const _Float16* krow = Kp + (nt * 16 + l15) * KD + 8 * lhalf;
      v8f acc = {};
#pragma unroll
      for (int k0 = 0; k0 < KD; k0 += 32) {
        v16h a   = afrag_f16(qrow + k0);
        v16h bfr = bfrag_f16(krow + k0);
        acc = WMMA_F16(a, bfr, acc);
      }
#pragma unroll
      for (int r = 0; r < 8; ++r)
        Sf[(r0 + r + 8 * lhalf) * JJ + nt * 16 + l15] = acc[r];
    }
  }

  // ---- Phase 2: wave-local row softmax + raw row-max ----
  for (int r = 0; r < 32; ++r) {
    int row = rbase + r;
    float v[8];
#pragma unroll
    for (int q = 0; q < 8; ++q) v[q] = Sf[row * JJ + lane + 32 * q];
    float m = v[0];
#pragma unroll
    for (int q = 1; q < 8; ++q) m = fmaxf(m, v[q]);
    for (int off = 16; off > 0; off >>= 1) m = fmaxf(m, __shfl_xor(m, off, 32));
    float p[8], l = 0.f;
#pragma unroll
    for (int q = 0; q < 8; ++q) { p[q] = __expf(SCALE_PER_SEQ * (v[q] - m)); l += p[q]; }
    for (int off = 16; off > 0; off >>= 1) l += __shfl_xor(l, off, 32);
    float inv = 1.0f / l;
    asm volatile("" ::: "memory");  // all f32 reads before f16 overwrite
#pragma unroll
    for (int q = 0; q < 8; ++q)
      Sh[row * 512 + lane + 32 * q] = (_Float16)(p[q] * inv);  // 1KB/row slot
    if (lane == 0) SMAX[(long)seq * II + row] = m;             // raw max
  }

  // ---- Phase 3: PSR = P @ V ----
  for (int mt = 0; mt < 2; ++mt) {
    int r0 = rbase + mt * 16;
    const _Float16* prow = Sh + (r0 + l15) * 512 + 8 * lhalf;
    for (int nt = 0; nt < 4; ++nt) {
      const _Float16* vrow = Vp + (nt * 16 + l15) * JJ + 8 * lhalf;
      v8f acc = {};
#pragma unroll
      for (int k0 = 0; k0 < JJ; k0 += 32) {
        v16h a   = afrag_f16(prow + k0);
        v16h bfr = bfrag_f16(vrow + k0);
        acc = WMMA_F16(a, bfr, acc);
      }
#pragma unroll
      for (int r = 0; r < 8; ++r)
        PSR[((long)seq * II + r0 + r + 8 * lhalf) * VD + nt * 16 + l15] = acc[r];
    }
  }
}

// ============================================================
// Kernel 3: level-2 softmax over t + weighted sum of PSR.
// grid = B*I blocks of 64 threads; thread = output channel v.
// ============================================================
__global__ __launch_bounds__(64) void combine_kernel(
    const float* __restrict__ SMAX, const float* __restrict__ PSR,
    float* __restrict__ out) {
  int bi = blockIdx.x;
  int b = bi >> 8, i = bi & 255;
  int v = threadIdx.x;
  const float* mrow = SMAX + (long)b * TT * II + i;      // stride II over t
  float mmax = mrow[0];
  for (int t = 1; t < TT; ++t) mmax = fmaxf(mmax, mrow[(long)t * II]);
  float acc = 0.f, lsum = 0.f;
  for (int t = 0; t < TT; ++t) {
    float w = __expf(SCALE_OVER_SEQS * (mrow[(long)t * II] - mmax));
    lsum += w;
    acc += w * PSR[(((long)(b * TT + t)) * II + i) * VD + v];
  }
  out[((long)b * II + i) * VD + v] = acc / lsum;
}

// ============================================================
extern "C" void kernel_launch(void* const* d_in, const int* in_sizes, int n_in,
                              void* d_out, int out_size, void* d_ws, size_t ws_size,
                              hipStream_t stream) {
  const float* input_seq = (const float*)d_in[0];
  const float* memory_x  = (const float*)d_in[1];
  const float* memory_y  = (const float*)d_in[2];
  const float* Wq = (const float*)d_in[3];
  const float* bq = (const float*)d_in[4];
  const float* Wk = (const float*)d_in[5];
  const float* bk = (const float*)d_in[6];
  const float* Wv = (const float*)d_in[7];
  const float* bv = (const float*)d_in[8];
  float* out = (float*)d_out;

  char* ws = (char*)d_ws;
  const size_t OFF_K    = 262144;                 // Qh : 8*256*64 f16 = 256 KB
  const size_t OFF_V    = OFF_K + 16777216;       // Kn : 512*256*64 f16 = 16 MB
  const size_t OFF_PSR  = OFF_V + 16777216;       // Vt : 512*64*256 f16 = 16 MB
  const size_t OFF_SMAX = OFF_PSR + 33554432;     // PSR: 512*256*64 f32 = 32 MB
  _Float16* Qh   = (_Float16*)ws;
  _Float16* Kn   = (_Float16*)(ws + OFF_K);
  _Float16* Vt   = (_Float16*)(ws + OFF_V);
  float*    PSR  = (float*)(ws + OFF_PSR);
  float*    SMAX = (float*)(ws + OFF_SMAX);       // 512*256 f32 = 512 KB

  // Projections: Q (2048 rows), K natural (131072 rows), V transposed
  proj_kernel<0><<<16,   256, 0, stream>>>(input_seq, Wq, bq, Qh);
  proj_kernel<0><<<1024, 256, 0, stream>>>(memory_x,  Wk, bk, Kn);
  proj_kernel<1><<<1024, 256, 0, stream>>>(memory_y,  Wv, bv, Vt);

  // Fused per-sequence attention (256 KB dynamic LDS per workgroup)
  attn_kernel<<<NSEQ, 256, 256 * 1024, stream>>>(Qh, Kn, Vt, PSR, SMAX);

  // Level-2 combine
  combine_kernel<<<BB * II, 64, 0, stream>>>(SMAX, PSR, out);
}
